// Quantize_17317308138093
// MI455X (gfx1250) — matrix operand order repaired and checked by reading, not compiled, not run on hardware
//
#include <hip/hip_runtime.h>

typedef __attribute__((ext_vector_type(16))) _Float16 v16h;
typedef __attribute__((ext_vector_type(8)))  _Float16 v8h;
typedef __attribute__((ext_vector_type(8)))  float    v8f;
typedef __attribute__((ext_vector_type(4)))  float    v4f;

#define BATCH   32
#define CH      64
#define HW      4096          // 64*64
#define NE      512
#define NROWS   131072        // BATCH*HW
#define ROWS_WG 128
#define NWG     (NROWS / ROWS_WG)   // 1024 workgroups, 32 per batch
#define BSTR    72            // padded halves per code row (144B: bank-conflict-free)

// ---------------------------------------------------------------------------
// Kernel 1: zero histogram counters + per-code squared norms ||e_j||^2
// grid: 2 x 256  (one thread per code)
// ---------------------------------------------------------------------------
__global__ void vq_prep_norms(const float* __restrict__ embed,
                              float* __restrict__ codeNorm,
                              unsigned* __restrict__ counts) {
    int j = blockIdx.x * blockDim.x + threadIdx.x;   // 0..511
    counts[j] = 0u;
    float s = 0.f;
#pragma unroll 8
    for (int c = 0; c < CH; ++c) {
        float e = embed[c * NE + j];
        s += e * e;
    }
    codeNorm[j] = s;
}

// ---------------------------------------------------------------------------
// Kernel 2: embed [64][512] fp32 -> embT [512][BSTR] f16, padded with zeros
// grid: 144 x 256  (512*72 = 36864 elements)
// ---------------------------------------------------------------------------
__global__ void vq_convert_embed(const float* __restrict__ embed,
                                 _Float16* __restrict__ embT) {
    int i = blockIdx.x * blockDim.x + threadIdx.x;   // 0..36863
    int col = i / BSTR;        // code
    int c   = i - col * BSTR;  // channel (or pad)
    embT[i] = (c < CH) ? (_Float16)embed[c * NE + col] : (_Float16)0.0f;
}

// ---------------------------------------------------------------------------
// Kernel 3: main VQ kernel. 1024 WGs x 256 threads (8 wave32).
// Codebook is async-DMA'd to LDS (ASYNCcnt path); x-tile staged as f16 A-tile.
// ---------------------------------------------------------------------------
__global__ __launch_bounds__(256)
void vq_main(const float* __restrict__ x,
             const float* __restrict__ embed,
             const _Float16* __restrict__ embT,
             const float* __restrict__ codeNorm,
             unsigned* __restrict__ counts,
             float* __restrict__ diffPartial,
             float* __restrict__ out) {
    __shared__ _Float16 Btile[NE * BSTR];      // 73728 B : f16 codebook, padded rows
    __shared__ float    cnLds[NE];             //  2048 B : ||e_j||^2
    __shared__ _Float16 Atile[ROWS_WG * CH];   // 16384 B : x tile (rows 128B apart)
    __shared__ int      idxArr[ROWS_WG];
    __shared__ float    wsum[8];

    const int tid = threadIdx.x;
    const int wg  = blockIdx.x;
    const int b   = wg >> 5;            // 32 WGs per batch
    const int hw0 = (wg & 31) << 7;     // 128 rows each
    const size_t xbase = (size_t)b * (CH * HW) + hw0;

    // ---- (1) kick off async DMA of codebook + norms into LDS (ASYNCcnt) ----
    {
        const unsigned bdst = (unsigned)(unsigned long long)(&Btile[0]);
        const unsigned long long bsrc = (unsigned long long)(const void*)embT;
        for (int e = tid; e < (NE * BSTR * 2) / 16; e += 256) {   // 4608 x 16B
            unsigned l = bdst + e * 16;
            unsigned long long g = bsrc + (unsigned long long)e * 16;
            asm volatile("global_load_async_to_lds_b128 %0, %1, off"
                         :: "v"(l), "v"(g) : "memory");
        }
        if (tid < 128) {                                          // 128 x 16B
            unsigned l = (unsigned)(unsigned long long)(&cnLds[0]) + tid * 16;
            unsigned long long g =
                (unsigned long long)(const void*)codeNorm + tid * 16;
            asm volatile("global_load_async_to_lds_b128 %0, %1, off"
                         :: "v"(l), "v"(g) : "memory");
        }
    }

    // ---- (2) stage x tile into LDS as f16 (overlaps with async DMA) ----
    for (int e = tid; e < 2048; e += 256) {          // 2048 float4 chunks
        int c = e >> 5;
        int r = (e & 31) << 2;
        v4f xv = *(const v4f*)(x + xbase + (size_t)c * HW + r);
        Atile[(r + 0) * CH + c] = (_Float16)xv[0];
        Atile[(r + 1) * CH + c] = (_Float16)xv[1];
        Atile[(r + 2) * CH + c] = (_Float16)xv[2];
        Atile[(r + 3) * CH + c] = (_Float16)xv[3];
    }
    __syncthreads();

    const int lane  = tid & 31;
    const int wave  = tid >> 5;
    const int half  = lane >> 4;   // 0 / 1
    const int ln    = lane & 15;
    const int rowA  = wave * 16 + ln;   // A-matrix M for this lane
    const int kbase = half * 8;         // K-half offset per documented A layout

    // ---- (3) A fragments (kc = 0: K 0..31, kc = 1: K 32..63) ----
    // 16-bit A layout: lanes 0-15 hold K {b..b+7, 16+b..16+b+7} with b=0; lanes 16-31 b=8.
    v16h afrag[2];
#pragma unroll
    for (int kc = 0; kc < 2; ++kc) {
        const _Float16* ap = &Atile[rowA * CH + kc * 32];
        v8h c0 = *(const v8h*)(ap + kbase);        // K = kc*32 + kbase .. +7
        v8h c1 = *(const v8h*)(ap + 16 + kbase);   // K = kc*32 + 16 + kbase .. +7
        v16h a;
#pragma unroll
        for (int i = 0; i < 8; ++i) { a[i] = c0[i]; a[8 + i] = c1[i]; }
        afrag[kc] = a;
    }

    // ---- (4) fence the async codebook DMA, then everyone can read Btile ----
    asm volatile("s_wait_asynccnt 0x0" ::: "memory");
    __syncthreads();

    // ---- (5) running argmin over 512 codes: dist = ||e||^2 - 2 f.e ----
    float minv[8];
    int   mini[8];
#pragma unroll
    for (int r = 0; r < 8; ++r) { minv[r] = 3.4e38f; mini[r] = 0; }

#pragma unroll 2
    for (int ct = 0; ct < 32; ++ct) {
        const int col = ct * 16 + ln;   // code column this lane owns (C-layout N)
        // 16-bit B layout: lanes 0-15 hold K {0..7, 16..23}, lanes 16-31 K {8..15, 24..31} (col = lane&15)
        const _Float16* bp = &Btile[col * BSTR + half * 8];
        v8h x0 = *(const v8h*)(bp);          // K 0..7   (+half*8)
        v8h x1 = *(const v8h*)(bp + 16);     // K 16..23 (+half*8)
        v8h y0 = *(const v8h*)(bp + 32);     // K 32..39 (+half*8)
        v8h y1 = *(const v8h*)(bp + 48);     // K 48..55 (+half*8)
        v16h bf0, bf1;
#pragma unroll
        for (int i = 0; i < 8; ++i) {
            bf0[i] = x0[i]; bf0[8 + i] = x1[i];
            bf1[i] = y0[i]; bf1[8 + i] = y1[i];
        }
        v8f acc = {};
        acc = __builtin_amdgcn_wmma_f32_16x16x32_f16(false, afrag[0], false, bf0,
                                                     (short)0, acc, false, false);
        acc = __builtin_amdgcn_wmma_f32_16x16x32_f16(false, afrag[1], false, bf1,
                                                     (short)0, acc, false, false);
        const float cn = cnLds[col];
#pragma unroll
        for (int r = 0; r < 8; ++r) {
            float d = __builtin_fmaf(-2.0f, acc[r], cn);
            if (d < minv[r]) { minv[r] = d; mini[r] = col; }
        }
    }

    // ---- (6) argmin across the 16 code lanes (xor tree, lowest-index tie-break) ----
#pragma unroll
    for (int off = 1; off < 16; off <<= 1) {
#pragma unroll
        for (int r = 0; r < 8; ++r) {
            float ov = __shfl_xor(minv[r], off, 32);
            int   oi = __shfl_xor(mini[r], off, 32);
            if (ov < minv[r] || (ov == minv[r] && oi < mini[r])) {
                minv[r] = ov; mini[r] = oi;
            }
        }
    }
    if (ln == 0) {
        // C-layout: VGPR r holds row M=r (lanes 0-15) / M=r+8 (lanes 16-31)
#pragma unroll
        for (int r = 0; r < 8; ++r) idxArr[wave * 16 + half * 8 + r] = mini[r];
    }
    __syncthreads();

    // ---- (7) gather exact fp32 code vectors, write straight-through output,
    //          accumulate commitment residual ----
    float local = 0.f;
    for (int e = tid; e < ROWS_WG * CH; e += 256) {
        int c = e >> 7, r = e & 127;
        int id = idxArr[r];
        float q  = embed[c * NE + id];
        size_t gi = xbase + (size_t)c * HW + r;
        float xv = x[gi];
        out[gi] = q;                       // xp + stopgrad(q - xp) == q
        float d = q - xv;
        local += d * d;
    }
#pragma unroll
    for (int off = 16; off > 0; off >>= 1) local += __shfl_xor(local, off, 32);
    if (lane == 0) wsum[wave] = local;
    __syncthreads();
    if (tid == 0) {
        float s = 0.f;
#pragma unroll
        for (int w = 0; w < 8; ++w) s += wsum[w];
        diffPartial[wg] = s;               // deterministic per-WG partial
    }
    if (tid < ROWS_WG) atomicAdd(&counts[idxArr[tid]], 1u);  // deterministic int atomics
}

// ---------------------------------------------------------------------------
// Kernel 4: per-batch diff + perplexity. 1 block x 256 threads.
// ---------------------------------------------------------------------------
__global__ void vq_final(const float* __restrict__ diffPartial,
                         const unsigned* __restrict__ counts,
                         float* __restrict__ outDiff,
                         float* __restrict__ outPerp) {
    __shared__ float ps[8];
    const int tid = threadIdx.x;

    if (tid < BATCH) {
        float s = 0.f;
#pragma unroll 8
        for (int i = 0; i < 32; ++i) s += diffPartial[tid * 32 + i];
        // COMMITMENT * mean over (H,W) then mean over C == sum / (HW*C)
        outDiff[tid] = s * (1.0f / (float)(HW * CH));
    }

    float ls = 0.f;
    for (int j = tid; j < NE; j += 256) {
        float p = (float)counts[j] * (1.0f / (float)NROWS);
        ls += p * logf(p + 1e-10f);
    }
#pragma unroll
    for (int off = 16; off > 0; off >>= 1) ls += __shfl_xor(ls, off, 32);
    if ((tid & 31) == 0) ps[tid >> 5] = ls;
    __syncthreads();
    if (tid == 0) {
        float t = 0.f;
#pragma unroll
        for (int w = 0; w < 8; ++w) t += ps[w];
        *outPerp = expf(-t);
    }
}

// ---------------------------------------------------------------------------
extern "C" void kernel_launch(void* const* d_in, const int* in_sizes, int n_in,
                              void* d_out, int out_size, void* d_ws, size_t ws_size,
                              hipStream_t stream) {
    (void)in_sizes; (void)n_in; (void)out_size; (void)ws_size;
    const float* x     = (const float*)d_in[0];   // [32,64,64,64] fp32
    const float* embed = (const float*)d_in[1];   // [64,512] fp32
    float* out = (float*)d_out;                   // 8388608 + 32 + 1 fp32

    char* ws = (char*)d_ws;                       // ~80 KB used
    _Float16* embT        = (_Float16*)(ws);           // 73728 B : f16 [512][72]
    float*    codeNorm    = (float*)(ws + 73728);      //  2048 B : ||e_j||^2
    unsigned* counts      = (unsigned*)(ws + 75776);   //  2048 B : histogram
    float*    diffPartial = (float*)(ws + 77824);      //  4096 B : per-WG sums

    vq_prep_norms   <<<2,   256, 0, stream>>>(embed, codeNorm, counts);
    vq_convert_embed<<<144, 256, 0, stream>>>(embed, embT);
    vq_main         <<<NWG, 256, 0, stream>>>(x, embed, embT, codeNorm,
                                              counts, diffPartial, out);
    vq_final        <<<1,   256, 0, stream>>>(diffPartial, counts,
                                              out + (size_t)NROWS * CH,
                                              out + (size_t)NROWS * CH + BATCH);
}
